// AMKPDModel_59450937311472
// MI455X (gfx1250) — matrix operand order repaired and checked by reading, not compiled, not run on hardware
//
#include <hip/hip_runtime.h>

// ---------------- model constants (from reference) ----------------
constexpr int cB = 2, cN = 1024, cD = 256, cH = 8, cK = 4;
constexpr int cI = 1024, cV = 32000, cCK = 3, cDH = 32;
constexpr int cMAXL = 8, cTRUNC = 4;
constexpr int cM = cB * cN;                 // 2048 rows of activations
constexpr size_t LOGITS_TOTAL = (size_t)(cMAXL - cTRUNC) * cB * cN * cV;

// ---------------- WMMA types ----------------
typedef __bf16 bf16;
typedef __attribute__((ext_vector_type(16))) __bf16 v16bf;
typedef __attribute__((ext_vector_type(8)))  float  v8f;

__device__ __forceinline__ unsigned pack2bf(float a, float b) {
  unsigned short x = __builtin_bit_cast(unsigned short, (bf16)a);
  unsigned short y = __builtin_bit_cast(unsigned short, (bf16)b);
  return (unsigned)x | ((unsigned)y << 16);
}

// K index inside a 16-bit A fragment (16x32), per CDNA5 ISA 7.12.2:
// lanes 0-15: VGPR0..3 -> K 0..7, VGPR4..7 -> K 16..23 ; lanes 16-31: +8
__device__ __forceinline__ int a_kidx(int e, int half) {
  int j = e >> 1;
  int base = (j < 4) ? (2 * j) : (2 * j + 8);
  return base + 8 * half + (e & 1);
}
// inverse map (value at k goes to lane-half (k>>3)&1, element (k&7)+8*(k>>4))

// ---------------- gfx1250 async global->LDS path (guarded) ----------------
#if defined(__has_builtin)
#if __has_builtin(__builtin_amdgcn_global_load_async_to_lds_b128)
#define HAVE_ASYNC_LDS 1
#endif
#if __has_builtin(__builtin_amdgcn_tensor_load_to_lds) && \
    __has_builtin(__builtin_amdgcn_s_wait_tensorcnt)
#define HAVE_TDM 1
#endif
#endif

typedef int v4i_gcc __attribute__((vector_size(16)));
#define ASG __attribute__((address_space(1)))
#define ASL __attribute__((address_space(3)))

#ifdef HAVE_ASYNC_LDS
__device__ __forceinline__ void async_copy16(const void* gsrc, void* ldst) {
  __builtin_amdgcn_global_load_async_to_lds_b128(
      (ASG v4i_gcc*)gsrc, (ASL v4i_gcc*)ldst, 0, 0);
}
__device__ __forceinline__ void async_wait0() {
#if __has_builtin(__builtin_amdgcn_s_wait_asynccnt)
  __builtin_amdgcn_s_wait_asynccnt(0);
#else
  asm volatile("s_wait_asynccnt 0x0" ::: "memory");
#endif
}
#endif

#ifdef HAVE_TDM
typedef unsigned u32x4_g __attribute__((vector_size(16)));
typedef int      i32x8_g __attribute__((vector_size(32)));
typedef int      i32x4_g __attribute__((vector_size(16)));
// One-shot TDM: contiguous tile of `elems` bf16 from global -> LDS.
// D# group0: count=1 | lds_addr | global_addr[56:0] | type=2
// D# group1: data_size=2B, tensor_dim0=tile_dim0=elems, dim1=1, stride=elems
__device__ __forceinline__ void tdm_load_tile(const bf16* gsrc, void* ldst,
                                              unsigned elems) {
  unsigned lds_off = (unsigned)(unsigned long long)(ASL bf16*)ldst;
  unsigned long long ga = (unsigned long long)gsrc;
  u32x4_g g0 = { 1u, lds_off, (unsigned)(ga & 0xFFFFFFFFu),
                 (unsigned)(((ga >> 32) & 0x1FFFFFFu) | (2u << 30)) };
  i32x8_g g1 = { 0x10000,                    // data_size = 2 bytes
                 (int)(elems << 16),         // tensor_dim0 (low 16 in hi half)
                 0x00010000,                 // tensor_dim1 = 1
                 (int)(elems << 16),         // tile_dim0
                 0, (int)elems, 0, 0 };      // tensor_dim0_stride
  i32x4_g z4 = { 0, 0, 0, 0 };
  i32x8_g z8 = { 0, 0, 0, 0, 0, 0, 0, 0 };
  __builtin_amdgcn_tensor_load_to_lds(g0, g1, z4, z4, z8, 0);
}
__device__ __forceinline__ void tdm_wait0() {
  __builtin_amdgcn_s_wait_tensorcnt(0);
}
#endif

// ---------------- generic WMMA GEMM: C = epi(A[M,Kd] @ W[Kd,Nc]) ----------------
// Weights are pre-permuted fragment-major: [kt][nt][lane 0..31][e 0..15] where
// value = W[kt*32 + 8*(lane>>4) + e + (e&8)][nt*16 + (lane&15)]  (512 elems / block).
constexpr int BM = 128, BN = 64, BK = 32;

constexpr int EPI_NONE  = 0;  // C[row*Nc+col] = x                      (fp32 out)
constexpr int EPI_ELU1  = 1;  // (b,h,n,dh) layout, elu(x)+1            (bf16 out)
constexpr int EPI_VT    = 2;  // (b,h,dh,n) transposed layout, raw      (bf16 out)
constexpr int EPI_AXPY  = 3;  // C = addend + softplus(dt)*x (in-place, fp32)
constexpr int EPI_ADD   = 4;  // C = addend + x              (in-place, fp32)

#define WMMA_BF16(A_, B_, C_, RA_) \
  __builtin_amdgcn_wmma_f32_16x16x32_bf16(false, (A_), false, (B_), (short)0, \
                                          (C_), (RA_), false)

template <int EPI>
__global__ __launch_bounds__(256, 2)
void gemm_wmma(const float* __restrict__ A, const bf16* __restrict__ Bp,
               float* __restrict__ C, const float* __restrict__ addend,
               const float* __restrict__ dt, int M, int Nc, int Kd) {
  __shared__ __align__(32) bf16 Asb[2][BM * BK];   // fragment-major, 2 x 8KB
  __shared__ __align__(32) bf16 Bsb[2][BK * BN];   // fragment-major, 2 x 4KB

  const int m0 = blockIdx.x * BM;
  const int n0 = blockIdx.y * BN;
  const int tid  = threadIdx.x;
  const int wave = tid >> 5;
  const int lane = tid & 31;
  const int half = lane >> 4;
  const int l16  = lane & 15;

  const int NT  = Nc >> 4;       // 16-col tiles per K row of permuted weights
  const int ntB = n0 >> 4;       // first n-tile of this block

  v8f acc[4];
#pragma unroll
  for (int t = 0; t < 4; ++t)
#pragma unroll
    for (int e = 0; e < 8; ++e) acc[t][e] = 0.0f;

  // A staging coords: thread handles row rA, 16 consecutive k at cA
  const int rA = tid >> 1, cA = (tid & 1) * 16;
  const int mA = rA & 15, rtA = rA >> 4;
  const int eBase = 8 * (cA >> 4);   // dest element base in fragment layout

  // ---- prologue: stage k-tile 0 into buffer 0 ----
#ifdef HAVE_TDM
  // Tensor Data Mover: one descriptor moves the whole 4KB fragment tile
  if (wave == 0) {
    tdm_load_tile(&Bp[((size_t)0 * NT + ntB) * 512], &Bsb[0][0], 2048u);
    tdm_wait0();
  }
#elif defined(HAVE_ASYNC_LDS)
  async_copy16(&Bp[((size_t)0 * NT + ntB) * 512 + tid * 8], &Bsb[0][tid * 8]);
#else
  *(uint4*)&Bsb[0][tid * 8] =
      *(const uint4*)&Bp[((size_t)0 * NT + ntB) * 512 + tid * 8];
#endif
  {
    const float4* Arow = (const float4*)(A + (size_t)(m0 + rA) * Kd + cA);
    float4 f0 = Arow[0], f1 = Arow[1], f2 = Arow[2], f3 = Arow[3];
    uint4 c0, c1;
    c0.x = pack2bf(f0.x, f0.y); c0.y = pack2bf(f0.z, f0.w);
    c0.z = pack2bf(f1.x, f1.y); c0.w = pack2bf(f1.z, f1.w);
    c1.x = pack2bf(f2.x, f2.y); c1.y = pack2bf(f2.z, f2.w);
    c1.z = pack2bf(f3.x, f3.y); c1.w = pack2bf(f3.z, f3.w);
    bf16* base = &Asb[0][rtA * 512];
    *(uint4*)&base[mA * 16 + eBase]        = c0;   // k=cA..cA+7  -> lane mA
    *(uint4*)&base[(mA + 16) * 16 + eBase] = c1;   // k=cA+8..+15 -> lane mA+16
  }
#if !defined(HAVE_TDM) && defined(HAVE_ASYNC_LDS)
  async_wait0();
#endif
  __syncthreads();

  const int ksteps = Kd / BK;
  for (int ks = 0; ks < ksteps; ++ks) {
    const int cur = ks & 1, nxt = cur ^ 1;

    // issue next B tile DMA before compute (overlap with WMMA)
    if (ks + 1 < ksteps) {
#ifdef HAVE_ASYNC_LDS
      async_copy16(&Bp[((size_t)(ks + 1) * NT + ntB) * 512 + tid * 8],
                   &Bsb[nxt][tid * 8]);
#else
      *(uint4*)&Bsb[nxt][tid * 8] =
          *(const uint4*)&Bp[((size_t)(ks + 1) * NT + ntB) * 512 + tid * 8];
#endif
    }
    // issue next A global loads before compute
    float4 f0, f1, f2, f3;
    if (ks + 1 < ksteps) {
      const float4* Arow =
          (const float4*)(A + (size_t)(m0 + rA) * Kd + (ks + 1) * BK + cA);
      f0 = Arow[0]; f1 = Arow[1]; f2 = Arow[2]; f3 = Arow[3];
    }
    if (ks + 2 < ksteps)   // global_prefetch_b8 two tiles ahead
      __builtin_prefetch(&A[(size_t)(m0 + rA) * Kd + (ks + 2) * BK + cA], 0, 1);

    // ---- compute from current buffers (contiguous fragment loads) ----
    v16bf afrag = *(const v16bf*)&Asb[cur][wave * 512 + lane * 16];
    {
      v16bf b0 = *(const v16bf*)&Bsb[cur][0 * 512 + lane * 16];
      v16bf b1 = *(const v16bf*)&Bsb[cur][1 * 512 + lane * 16];
      v16bf b2 = *(const v16bf*)&Bsb[cur][2 * 512 + lane * 16];
      v16bf b3 = *(const v16bf*)&Bsb[cur][3 * 512 + lane * 16];
      acc[0] = WMMA_BF16(afrag, b0, acc[0], false);
      acc[1] = WMMA_BF16(afrag, b1, acc[1], true);   // reuse-A hint
      acc[2] = WMMA_BF16(afrag, b2, acc[2], true);
      acc[3] = WMMA_BF16(afrag, b3, acc[3], true);
    }

    // ---- stage next A tile, wait DMA, hand over ----
    if (ks + 1 < ksteps) {
      uint4 c0, c1;
      c0.x = pack2bf(f0.x, f0.y); c0.y = pack2bf(f0.z, f0.w);
      c0.z = pack2bf(f1.x, f1.y); c0.w = pack2bf(f1.z, f1.w);
      c1.x = pack2bf(f2.x, f2.y); c1.y = pack2bf(f2.z, f2.w);
      c1.z = pack2bf(f3.x, f3.y); c1.w = pack2bf(f3.z, f3.w);
      bf16* base = &Asb[nxt][rtA * 512];
      *(uint4*)&base[mA * 16 + eBase]        = c0;
      *(uint4*)&base[(mA + 16) * 16 + eBase] = c1;
#ifdef HAVE_ASYNC_LDS
      async_wait0();
#endif
    }
    __syncthreads();
  }

  float dtscale = 0.0f;
  if (EPI == EPI_AXPY) {
    float d = dt[0];
    dtscale = (d > 20.0f) ? d : log1pf(__expf(d));  // softplus
  }
  bf16* Cb = (bf16*)C;

#pragma unroll
  for (int t = 0; t < 4; ++t) {
#pragma unroll
    for (int r = 0; r < 8; ++r) {
      int row = m0 + wave * 16 + r + 8 * half;
      int col = n0 + t * 16 + l16;
      float x = acc[t][r];
      if constexpr (EPI == EPI_NONE) {
        C[(size_t)row * Nc + col] = x;
      } else if constexpr (EPI == EPI_ELU1) {
        int b = row >> 10, n = row & (cN - 1);
        int h = col >> 5,  dh = col & (cDH - 1);
        float p = (x > 0.0f) ? (x + 1.0f) : __expf(x);   // elu(x)+1
        Cb[(((size_t)(b * cH + h)) * cN + n) * cDH + dh] = (bf16)p;
      } else if constexpr (EPI == EPI_VT) {
        int b = row >> 10, n = row & (cN - 1);
        int h = col >> 5,  dh = col & (cDH - 1);
        Cb[(((size_t)(b * cH + h)) * cDH + dh) * cN + n] = (bf16)x;
      } else if constexpr (EPI == EPI_AXPY) {
        C[(size_t)row * Nc + col] =
            addend[(size_t)row * Nc + col] + dtscale * x;
      } else if constexpr (EPI == EPI_ADD) {
        C[(size_t)row * Nc + col] = addend[(size_t)row * Nc + col] + x;
      }
    }
  }
}

// ---------------- streaming linear attention (flash-style, no NxN tensor) ----------------
// m = (W @ v) / (rowsum(W)+1) - v ,  W = relu(pq·pk^T)^2, per (b,h).
// pq/pk: bf16 (b,h,n,dh);  v: bf16 transposed (b,h,dh,n).
__global__ __launch_bounds__(128, 2)
void attn_kernel(const bf16* __restrict__ pq, const bf16* __restrict__ pk,
                 const bf16* __restrict__ vv, float* __restrict__ mOut) {
  __shared__ __align__(32) bf16 Ss[4][32 * 16];  // per-wave S in fragment order
  const int bh   = blockIdx.x;              // b*H+h
  const int tile = blockIdx.y;              // N/64 row tiles
  const int tid  = threadIdx.x;
  const int wave = tid >> 5, lane = tid & 31;
  const int half = lane >> 4, l16 = lane & 15;
  const int q0   = tile * 64 + wave * 16;

  const bf16* pqB = pq + (size_t)bh * cN * cDH;
  const bf16* pkB = pk + (size_t)bh * cN * cDH;
  const bf16* vB  = vv + (size_t)bh * cN * cDH;   // [dh][n]

  // pq A-fragment (16x32, K = DH) lives in registers for the whole pass
  v16bf aq;
#pragma unroll
  for (int e = 0; e < 16; ++e)
    aq[e] = pqB[(size_t)(q0 + l16) * cDH + e + 16 * half];

  v8f acc0, acc1;
  float rs[8];
#pragma unroll
  for (int e = 0; e < 8; ++e) { acc0[e] = 0.0f; acc1[e] = 0.0f; rs[e] = 0.0f; }

  // fragment-slot coords for the S store (consumer mapping)
  const int eS = l16 & 7;
  const int hS = (l16 >> 3) & 1;

  for (int kb = 0; kb < cN / 32; ++kb) {
    const int key0 = kb * 32;
    // S = pq @ pk^T  (two 16x16 output tiles covering 32 keys)
    v16bf bk0, bk1;
#pragma unroll
    for (int e = 0; e < 16; ++e) {
      bk0[e] = pkB[(size_t)(key0 + l16) * cDH + e + 16 * half];
      bk1[e] = pkB[(size_t)(key0 + 16 + l16) * cDH + e + 16 * half];
    }
    v8f z;
#pragma unroll
    for (int e = 0; e < 8; ++e) z[e] = 0.0f;
    v8f s0 = WMMA_BF16(aq, bk0, z, false);
    v8f s1 = WMMA_BF16(aq, bk1, z, true);      // reuse-A hint
    // relu^2 + rowsum; store S directly into the consumer fragment slots
    bf16* S = &Ss[wave][0];
#pragma unroll
    for (int r = 0; r < 8; ++r) {
      float x0 = s0[r] > 0.0f ? s0[r] : 0.0f; x0 *= x0;   // col = l16
      float x1 = s1[r] > 0.0f ? s1[r] : 0.0f; x1 *= x1;   // col = 16 + l16
      rs[r] += x0 + x1;
      int laneC = (r + 8 * half) + 16 * hS;   // consumer lane for both cols
      S[laneC * 16 + eS]     = (bf16)x0;
      S[laneC * 16 + eS + 8] = (bf16)x1;
    }
    __syncthreads();
    // A-fragment of S: contiguous 32B per lane
    v16bf as = *(const v16bf*)&S[lane * 16];
    // v B-fragments from transposed layout: k = e + (e&8) + 8*half (two 8-key runs)
    v16bf bv0, bv1;
#pragma unroll
    for (int e = 0; e < 16; ++e) {
      bv0[e] = vB[(size_t)l16 * cN + key0 + e + (e & 8) + 8 * half];
      bv1[e] = vB[(size_t)(16 + l16) * cN + key0 + e + (e & 8) + 8 * half];
    }
    acc0 = WMMA_BF16(as, bv0, acc0, false);
    acc1 = WMMA_BF16(as, bv1, acc1, true);     // reuse-A hint
    __syncthreads();
  }

  // reduce rowsums across the 16-lane half-groups (wave32 xor shuffles)
#pragma unroll
  for (int r = 0; r < 8; ++r) {
    float s = rs[r];
    s += __shfl_xor(s, 1, 32);
    s += __shfl_xor(s, 2, 32);
    s += __shfl_xor(s, 4, 32);
    s += __shfl_xor(s, 8, 32);
    rs[r] = s;
  }

  const int b = bh >> 3, h = bh & 7;
#pragma unroll
  for (int r = 0; r < 8; ++r) {
    int n = q0 + r + 8 * half;
    float inv = 1.0f / (rs[r] + 1.0f);
    float c0 = acc0[r] * inv - (float)vB[(size_t)l16 * cN + n];
    float c1 = acc1[r] * inv - (float)vB[(size_t)(16 + l16) * cN + n];
    mOut[((size_t)b * cN + n) * cD + h * cDH + l16]      = c0;
    mOut[((size_t)b * cN + n) * cD + h * cDH + 16 + l16] = c1;
  }
}

// ---------------- small elementwise / reduction kernels ----------------
__global__ void embed_kernel(const int* __restrict__ ids,
                             const float* __restrict__ emb,
                             const float* __restrict__ pos,
                             float* __restrict__ out) {
  int row = blockIdx.x, i = threadIdx.x;
  int n = row & (cN - 1);
  out[(size_t)row * cD + i] =
      emb[(size_t)ids[row] * cD + i] + pos[(size_t)n * cD + i];
}

__global__ void ln_kernel(const float* __restrict__ x, const float* __restrict__ w,
                          const float* __restrict__ bia, const float* __restrict__ addx,
                          float* __restrict__ out) {
  __shared__ float red[cD];
  int row = blockIdx.x, i = threadIdx.x;
  float xv = x[(size_t)row * cD + i];
  red[i] = xv; __syncthreads();
  for (int s = cD / 2; s > 0; s >>= 1) { if (i < s) red[i] += red[i + s]; __syncthreads(); }
  float mu = red[0] * (1.0f / cD);
  __syncthreads();
  float dv = xv - mu;
  red[i] = dv * dv; __syncthreads();
  for (int s = cD / 2; s > 0; s >>= 1) { if (i < s) red[i] += red[i + s]; __syncthreads(); }
  float var = red[0] * (1.0f / cD);
  float y = dv * rsqrtf(var + 1e-5f) * w[i] + bia[i];
  if (addx) y += addx[(size_t)row * cD + i];
  out[(size_t)row * cD + i] = y;
}

__global__ void copy_f32(const float* __restrict__ a, float* __restrict__ b, size_t n) {
  size_t i = (size_t)blockIdx.x * blockDim.x + threadIdx.x;
  if (i < n) b[i] = a[i];
}

// fp32 W[Kd][Nc] -> bf16 fragment-major [kt][nt][lane][e]; thread per (kt,nt,lane)
__global__ void cvt_wfrag(const float* __restrict__ W, bf16* __restrict__ Bp,
                          int Nc, int Kd) {
  int tidg = blockIdx.x * blockDim.x + threadIdx.x;
  int NT = Nc >> 4;
  int total = (Kd >> 5) * NT * 32;
  if (tidg >= total) return;
  int lane = tidg & 31;
  int blk  = tidg >> 5;                 // kt*NT + nt
  int kt = blk / NT, nt = blk - kt * NT;
  int n = nt * 16 + (lane & 15);
  int kbase = kt * 32 + 8 * (lane >> 4);
  const float* col = W + (size_t)kbase * Nc + n;
  float v[16];
#pragma unroll
  for (int e = 0; e < 16; ++e) v[e] = col[(size_t)(e + (e & 8)) * Nc];
  uint4 c0, c1;
  c0.x = pack2bf(v[0], v[1]);   c0.y = pack2bf(v[2], v[3]);
  c0.z = pack2bf(v[4], v[5]);   c0.w = pack2bf(v[6], v[7]);
  c1.x = pack2bf(v[8], v[9]);   c1.y = pack2bf(v[10], v[11]);
  c1.z = pack2bf(v[12], v[13]); c1.w = pack2bf(v[14], v[15]);
  uint4* dst = (uint4*)&Bp[(size_t)blk * 512 + lane * 16];
  dst[0] = c0; dst[1] = c1;
}

__global__ void silu_gate(const float* __restrict__ GU, float* __restrict__ Hf) {
  size_t idx = (size_t)blockIdx.x * blockDim.x + threadIdx.x;  // B*N*INNER
  size_t row = idx / cI; int i = (int)(idx % cI);
  float g = GU[row * (2 * cI) + i];
  float u = GU[row * (2 * cI) + cI + i];
  Hf[idx] = (g / (1.0f + __expf(-g))) * u;
}

__global__ void dwconv3(const float* __restrict__ Hf, const float* __restrict__ cw,
                        float* __restrict__ out) {
  size_t idx = (size_t)blockIdx.x * blockDim.x + threadIdx.x;  // B*N*INNER
  size_t row = idx / cI; int i = (int)(idx % cI);
  int b = (int)(row >> 10), n = (int)(row & (cN - 1));
  float y = 0.0f;
#pragma unroll
  for (int t = 0; t < cCK; ++t) {
    int nn = n + t - 1;
    if (nn >= 0 && nn < cN)
      y += Hf[((size_t)(b * cN + nn)) * cI + i] * cw[(size_t)i * cCK + t];
  }
  out[idx] = y;
}

__global__ void halt_kernel(const float* __restrict__ Qn, const float* __restrict__ hw,
                            const float* __restrict__ hb, float* __restrict__ out) {
  __shared__ float red[cD];
  int b = blockIdx.x, d = threadIdx.x;
  float s = 0.0f;
  for (int n = 0; n < cN; ++n) s += Qn[((size_t)(b * cN + n)) * cD + d];
  red[d] = (s * (1.0f / cN)) * hw[d];
  __syncthreads();
  for (int st = cD / 2; st > 0; st >>= 1) { if (d < st) red[d] += red[d + st]; __syncthreads(); }
  if (d == 0) out[b] = 1.0f / (1.0f + __expf(-(red[0] + hb[0])));
}

// ---------------- host orchestration ----------------
extern "C" void kernel_launch(void* const* d_in, const int* in_sizes, int n_in,
                              void* d_out, int out_size, void* d_ws, size_t ws_size,
                              hipStream_t stream) {
  (void)in_sizes; (void)n_in; (void)out_size; (void)ws_size;
  const int*   ids   = (const int*)  d_in[0];
  const float* emb   = (const float*)d_in[1];
  const float* pos   = (const float*)d_in[2];
  const float* in_w  = (const float*)d_in[3];
  const float* in_b  = (const float*)d_in[4];
  const float* Wq    = (const float*)d_in[5];
  const float* Wk    = (const float*)d_in[6];
  const float* Wv    = (const float*)d_in[7];
  const float* Wo    = (const float*)d_in[8];
  const float* dts   = (const float*)d_in[9];
  const float* Wup   = (const float*)d_in[10];
  const float* cw    = (const float*)d_in[11];
  const float* Wd    = (const float*)d_in[12];
  const float* n1w   = (const float*)d_in[13];
  const float* n1b   = (const float*)d_in[14];
  const float* n2w   = (const float*)d_in[15];
  const float* n2b   = (const float*)d_in[16];
  const float* fin_w = (const float*)d_in[17];
  const float* fin_b = (const float*)d_in[18];
  const float* hlt_w = (const float*)d_in[19];
  const float* hlt_b = (const float*)d_in[20];
  const float* lm_w  = (const float*)d_in[21];
  float* fout = (float*)d_out;

  // workspace carving
  char* wsb = (char*)d_ws;
  size_t off = 0;
  auto carve = [&](size_t bytes) -> void* {
    void* p = wsb + off;
    off += (bytes + 255) & ~(size_t)255;
    return p;
  };
  const size_t szQKVO = (size_t)cK * cD * cD;
  const size_t szUP   = (size_t)cK * cD * 2 * cI;
  const size_t szDN   = (size_t)cK * cI * cD;
  const size_t szLM   = (size_t)cD * cV;
  bf16* wq_bf = (bf16*)carve(szQKVO * 2);
  bf16* wk_bf = (bf16*)carve(szQKVO * 2);
  bf16* wv_bf = (bf16*)carve(szQKVO * 2);
  bf16* wo_bf = (bf16*)carve(szQKVO * 2);
  bf16* up_bf = (bf16*)carve(szUP * 2);
  bf16* dn_bf = (bf16*)carve(szDN * 2);
  bf16* lm_bf = (bf16*)carve(szLM * 2);
  const size_t szAct = (size_t)cM * cD;       // 2048*256
  float* X   = (float*)carve(szAct * 4);
  float* Q   = (float*)carve(szAct * 4);
  float* Hc  = (float*)carve(szAct * 4);
  bf16*  pqA = (bf16*)carve(szAct * 2);       // bf16 head-layout activations
  bf16*  pkA = (bf16*)carve(szAct * 2);
  bf16*  vA  = (bf16*)carve(szAct * 2);       // transposed (b,h,dh,n)
  float* mA  = (float*)carve(szAct * 4);
  float* ln2 = (float*)carve(szAct * 4);
  float* Qn  = (float*)carve(szAct * 4);
  float* GU  = (float*)carve((size_t)cM * 2 * cI * 4);
  float* Hf  = (float*)carve((size_t)cM * cI * 4);
  float* Hc2 = (float*)carve((size_t)cM * cI * 4);

  // weight conversion into WMMA-fragment-major bf16 layout
  auto cvtW = [&](const float* s, bf16* d, int Nc_, int Kd_) {
    int total = (Kd_ >> 5) * (Nc_ >> 4) * 32;
    cvt_wfrag<<<dim3((unsigned)((total + 255) / 256)), 256, 0, stream>>>(
        s, d, Nc_, Kd_);
  };
  for (int k = 0; k < cK; ++k) {
    cvtW(Wq + (size_t)k * cD * cD, wq_bf + (size_t)k * cD * cD, cD, cD);
    cvtW(Wk + (size_t)k * cD * cD, wk_bf + (size_t)k * cD * cD, cD, cD);
    cvtW(Wv + (size_t)k * cD * cD, wv_bf + (size_t)k * cD * cD, cD, cD);
    cvtW(Wo + (size_t)k * cD * cD, wo_bf + (size_t)k * cD * cD, cD, cD);
    cvtW(Wup + (size_t)k * cD * 2 * cI, up_bf + (size_t)k * cD * 2 * cI, 2 * cI, cD);
    cvtW(Wd + (size_t)k * cI * cD, dn_bf + (size_t)k * cI * cD, cD, cI);
  }
  cvtW(lm_w, lm_bf, cV, cD);

  // X = ln(emb[ids] + pos); Q = X
  embed_kernel<<<cM, cD, 0, stream>>>(ids, emb, pos, Hc);
  ln_kernel<<<cM, cD, 0, stream>>>(Hc, in_w, in_b, nullptr, X);
  copy_f32<<<(unsigned)((szAct + 255) / 256), 256, 0, stream>>>(X, Q, szAct);

  const dim3 gemmBlk(256);
  const dim3 gD(cM / BM, cD / 64);        // Nc = 256
  const dim3 gUP(cM / BM, (2 * cI) / 64); // Nc = 2048
  const dim3 gLM(cM / BM, cV / 64);       // Nc = 32000
  const dim3 attG(cB * cH, cN / 64);
  const unsigned elemI = (unsigned)(((size_t)cM * cI + 255) / 256);

  for (int l = 0; l < cMAXL; ++l) {
    for (int k = 0; k < cK; ++k) {
      // Hc = ln(Q, n1) + X
      ln_kernel<<<cM, cD, 0, stream>>>(Q, n1w + k * cD, n1b + k * cD, X, Hc);
      // projections (fused elu+1 for q,k; transposed bf16 layout for v)
      gemm_wmma<EPI_ELU1><<<gD, gemmBlk, 0, stream>>>(
          Hc, wq_bf + (size_t)k * cD * cD, (float*)pqA, nullptr, nullptr, cM, cD, cD);
      gemm_wmma<EPI_ELU1><<<gD, gemmBlk, 0, stream>>>(
          Hc, wk_bf + (size_t)k * cD * cD, (float*)pkA, nullptr, nullptr, cM, cD, cD);
      gemm_wmma<EPI_VT><<<gD, gemmBlk, 0, stream>>>(
          Hc, wv_bf + (size_t)k * cD * cD, (float*)vA, nullptr, nullptr, cM, cD, cD);
      // streaming quadratic-kernel attention
      attn_kernel<<<attG, 128, 0, stream>>>(pqA, pkA, vA, mA);
      // Q = Q + softplus(dt) * (m @ Wo)
      gemm_wmma<EPI_AXPY><<<gD, gemmBlk, 0, stream>>>(
          mA, wo_bf + (size_t)k * cD * cD, Q, Q, dts + k, cM, cD, cD);
      // MLP
      ln_kernel<<<cM, cD, 0, stream>>>(Q, n2w + k * cD, n2b + k * cD, nullptr, ln2);
      gemm_wmma<EPI_NONE><<<gUP, gemmBlk, 0, stream>>>(
          ln2, up_bf + (size_t)k * cD * 2 * cI, GU, nullptr, nullptr, cM, 2 * cI, cD);
      silu_gate<<<elemI, 256, 0, stream>>>(GU, Hf);
      dwconv3<<<elemI, 256, 0, stream>>>(Hf, cw + (size_t)k * cI * cCK, Hc2);
      gemm_wmma<EPI_ADD><<<gD, gemmBlk, 0, stream>>>(
          Hc2, dn_bf + (size_t)k * cI * cD, Q, Q, nullptr, cM, cD, cI);
    }
    if (l >= cTRUNC) {
      const int li = l - cTRUNC;
      ln_kernel<<<cM, cD, 0, stream>>>(Q, fin_w, fin_b, nullptr, Qn);
      halt_kernel<<<cB, cD, 0, stream>>>(Qn, hlt_w, hlt_b,
                                         fout + LOGITS_TOTAL + (size_t)li * cB);
      gemm_wmma<EPI_NONE><<<gLM, gemmBlk, 0, stream>>>(
          Qn, lm_bf, fout + (size_t)li * cB * cN * cV, nullptr, nullptr, cM, cV, cD);
    }
  }
}
